// FPConv6x6_BaseBlock_90082644066729
// MI455X (gfx1250) — matrix operand order repaired and checked by reading, not compiled, not run on hardware
//
#include <hip/hip_runtime.h>
#include <hip/hip_bf16.h>

typedef __attribute__((ext_vector_type(16))) _Float16 v16h;
typedef __attribute__((ext_vector_type(8)))  _Float16 v8h;
typedef __attribute__((ext_vector_type(8)))  float    v8f;

#define ALPHA   0.2f
#define RAD     0.15f
#define NPTS    8192
#define NQ      2048
#define CINCH   128
#define COCH    256
#define NS      32
#define NB      4
#define BPTOT   (NB*NQ)      // 8192 samples

// gfT row pitch (f16): 40 -> row base 80B (16B aligned), lane-row reads hit
// 16 distinct LDS banks; transpose scatter is lane-contiguous (conflict-free).
#define GFP     40
// img: 37 rows per sample (row 36 = dump row for the 48->36 M padding)
#define IMGP    (37*CINCH)   // 4736 f16 per sample

__device__ __forceinline__ float lrelu(float x){ return x > 0.f ? x : ALPHA*x; }

__device__ __forceinline__ unsigned ballot32(bool p){
#if __has_builtin(__builtin_amdgcn_ballot_w32)
  return __builtin_amdgcn_ballot_w32(p);
#else
  return (unsigned)__ballot(p);
#endif
}

__device__ __forceinline__ v8f wmma16(v16h a, v16h b, v8f c){
  // D = A(16x32 f16) * B(32x16 f16) + C(16x16 f32)
  return __builtin_amdgcn_wmma_f32_16x16x32_f16(false, a, false, b, (short)0, c, false, false);
}

// Pack two 8-element runs into one 16-element fragment.
//  A-fragment: runs at +0 / +16 f16 (K-runs {lh*8..}, {16+lh*8..})
//  B-fragment: runs at +0 / +16 f16 (same K-runs as A)
__device__ __forceinline__ v16h pack16(const _Float16* lo, const _Float16* hi){
  v8h a = *(const v8h*)lo;
  v8h b = *(const v8h*)hi;
  v16h r;
#pragma unroll
  for (int i=0;i<8;i++){ r[i]=a[i]; r[i+8]=b[i]; }
  return r;
}

// ---------------------------------------------------------------------------
// Tensor Data Mover: 1-D contiguous tile load, Global -> LDS.
//  - ROCm 7.2 (clang-22): probe-verified 5-arg builtin.
//  - otherwise: inline asm "tensor_load_to_lds" (portable per CDNA5_HIP.md);
//    D# groups are wave-uniform, bound to aligned SGPR tuples via "s".
// ---------------------------------------------------------------------------
typedef unsigned int u32x4 __attribute__((ext_vector_type(4)));
typedef int          i32x8 __attribute__((ext_vector_type(8)));
typedef int          i32x4 __attribute__((ext_vector_type(4)));

#if defined(__has_builtin)
#if __has_builtin(__builtin_amdgcn_tensor_load_to_lds) && (__clang_major__ < 23)
#define TDM_BUILTIN 1
#endif
#endif

__device__ __forceinline__ void tdm_load_1d(unsigned lds_byte_off,
                                            const void* gsrc,
                                            unsigned nelem /*f16 elems*/){
  unsigned long long ga = (unsigned long long)gsrc;
  u32x4 g0;
  g0[0] = 1u;                                           // count=1 (valid user D#)
  g0[1] = lds_byte_off;                                 // lds_addr (bytes)
  g0[2] = (unsigned)(ga & 0xffffffffu);                 // global_addr[31:0]
  g0[3] = ((unsigned)(ga >> 32) & 0x01ffffffu)          // global_addr[56:32]
        | (2u << 30);                                   // type = 2 ("image")
  i32x8 g1;
  g1[0] = (int)(1u << 16);                              // data_size = 1 -> 2B
  g1[1] = (int)((nelem & 0xffffu) << 16);               // tensor_dim0[15:0]
  g1[2] = (int)((nelem >> 16) | (1u << 16));            // tensor_dim0 hi | tensor_dim1=1
  g1[3] = (int)(nelem << 16);                           // tile_dim0
  g1[4] = 1;                                            // tile_dim1 = 1
  g1[5] = (int)nelem;                                   // tensor_dim0_stride
  g1[6] = 0;
  g1[7] = 0;
  i32x4 z = {0,0,0,0};                                  // groups 2/3: dims unused
#ifdef TDM_BUILTIN
  __builtin_amdgcn_tensor_load_to_lds(g0, g1, z, z, 0);
#else
  asm volatile("tensor_load_to_lds %0, %1, %2, %3"
               :: "s"(g0), "s"(g1), "s"(z), "s"(z)
               : "memory");
#endif
}

__device__ __forceinline__ void tdm_wait(){
#if defined(__has_builtin) && __has_builtin(__builtin_amdgcn_s_wait_tensorcnt)
  __builtin_amdgcn_s_wait_tensorcnt(0);
#else
  asm volatile("s_wait_tensorcnt 0x0" ::: "memory");
#endif
}

// ---------------------------------------------------------------------------
// K0: features (B,C,N) f32 -> featT (B,N,C) f16   (tiled LDS transpose)
// ---------------------------------------------------------------------------
__global__ __launch_bounds__(256) void k_featT(const float* __restrict__ f,
                                               _Float16* __restrict__ featT){
  __shared__ float tile[32][33];
  int b = blockIdx.z, n0 = blockIdx.x*32, c0 = blockIdx.y*32;
  int tx = threadIdx.x, ty = threadIdx.y;
#pragma unroll
  for (int i=ty;i<32;i+=8)
    tile[i][tx] = f[((size_t)b*CINCH + (c0+i))*NPTS + n0 + tx];
  __syncthreads();
#pragma unroll
  for (int i=ty;i<32;i+=8)
    featT[((size_t)b*NPTS + n0 + i)*CINCH + c0 + tx] = (_Float16)tile[tx][i];
}

// ---------------------------------------------------------------------------
// K1: pack conv weights into WMMA B-operand order (f16)
//   element e = ((kk*NT + nt)*32 + lane)*16 + j ; K = kk*32 + (lane>>4)*8 + (j&7) + (j>>3)*16
//   N = nt*16 + (lane&15); K ordering = tap*128 + cin
// ---------------------------------------------------------------------------
__global__ __launch_bounds__(256) void k_wprep(const float* __restrict__ c1w,
                                               const float* __restrict__ c2w,
                                               const float* __restrict__ c3w,
                                               _Float16* __restrict__ w1B,
                                               _Float16* __restrict__ w2B,
                                               _Float16* __restrict__ w3B){
  int e = blockIdx.x*256 + threadIdx.x;
  if (e < 36*8*32*16){                       // conv1/conv2: K=1152, N=128
    int j    =  e        & 15;
    int lane = (e >> 4)  & 31;
    int nt   = (e >> 9)  & 7;
    int kk   =  e >> 12;
    int k   = kk*32 + ((lane>>4)<<3) + (j & 7) + ((j>>3)<<4);    // 0..1151
    int n   = nt*16 + (lane&15);             // 0..127
    int tap = k >> 7, cin = k & 127;
    int kh  = tap/3, kw = tap - kh*3;
    size_t src = (((size_t)n*CINCH + cin)*3 + kh)*3 + kw;
    w1B[e] = (_Float16)c1w[src];
    w2B[e] = (_Float16)c2w[src];
  }
  if (e < 16*16*32*16){                      // conv3: K=512, N=256
    int j    =  e        & 15;
    int lane = (e >> 4)  & 31;
    int nt   = (e >> 9)  & 15;
    int kk   =  e >> 13;
    int k   = kk*32 + ((lane>>4)<<3) + (j & 7) + ((j>>3)<<4);    // 0..511
    int n   = nt*16 + (lane&15);             // 0..255
    int tap = k >> 7, cin = k & 127;
    int kh  = tap >> 1, kw = tap & 1;
    w3B[e] = (_Float16)c3w[(((size_t)n*CINCH + cin)*2 + kh)*2 + kw];
  }
}

// ---------------------------------------------------------------------------
// K2: ball query (one wave per query). Ordered compaction via ballot+popc.
// ---------------------------------------------------------------------------
__global__ __launch_bounds__(256) void k_ballquery(const float* __restrict__ xyz,
                                                   const float* __restrict__ nxyz,
                                                   int* __restrict__ idxg,
                                                   float* __restrict__ gout){
  __shared__ float pts[512*3];
  __shared__ int   idxb[8][NS];
  __shared__ float gxb[8][NS][3];
  int tid = threadIdx.x, w = tid>>5, l = tid&31;
  int bp = blockIdx.x*8 + w;
  int b  = bp >> 11;
  float qx = nxyz[bp*3+0], qy = nxyz[bp*3+1], qz = nxyz[bp*3+2];
  const float r2 = RAD*RAD, invR = 1.0f/RAD;
  int cnt = 0;
  for (int ch=0; ch<16; ++ch){
    __syncthreads();
    for (int i=tid;i<1536;i+=256)
      pts[i] = xyz[((size_t)b*NPTS + ch*512)*3 + i];
    __syncthreads();
    if (cnt < NS){
      for (int it=0; it<16; ++it){
        int li = it*32 + l;
        float dx = pts[li*3+0]-qx, dy = pts[li*3+1]-qy, dz = pts[li*3+2]-qz;
        float d2 = dx*dx + dy*dy + dz*dz;
        bool in = d2 < r2;
        unsigned mask = ballot32(in);
        int slot = cnt + __popc(mask & ((1u<<l)-1u));
        if (in && slot < NS){
          idxb[w][slot]   = ch*512 + li;
          gxb[w][slot][0] = dx*invR;
          gxb[w][slot][1] = dy*invR;
          gxb[w][slot][2] = dz*invR;
        }
        cnt += __popc(mask);
        if (cnt >= NS) break;
      }
    }
  }
  __syncthreads();
  int c = cnt < NS ? cnt : NS;
  int id; float g0,g1,g2;
  if (l < c){
    id = idxb[w][l]; g0 = gxb[w][l][0]; g1 = gxb[w][l][1]; g2 = gxb[w][l][2];
  } else if (c > 0){
    id = idxb[w][0]; g0 = gxb[w][0][0]; g1 = gxb[w][0][1]; g2 = gxb[w][0][2];
  } else {
    id = NPTS-1;
    g0 = (xyz[((size_t)b*NPTS + NPTS-1)*3+0]-qx)*invR;
    g1 = (xyz[((size_t)b*NPTS + NPTS-1)*3+1]-qy)*invR;
    g2 = (xyz[((size_t)b*NPTS + NPTS-1)*3+2]-qz)*invR;
  }
  idxg[(size_t)bp*NS + l]          = id;
  gout[((size_t)bp*NS + l)*3 + 0]  = g0;
  gout[((size_t)bp*NS + l)*3 + 1]  = g1;
  gout[((size_t)bp*NS + l)*3 + 2]  = g2;
}

// ---------------------------------------------------------------------------
// K3: per-(b,p,s) MLP + two-sided normalization. Lane = sample s.
//     Output stored transposed: wtsT[bp][m][s] f16, m padded to 48 (rows 36..47 = 0)
// ---------------------------------------------------------------------------
__global__ __launch_bounds__(256) void k_mlp(const float* __restrict__ gxyz,
    const float* __restrict__ pw1, const float* __restrict__ pb1,
    const float* __restrict__ pw2, const float* __restrict__ pb2,
    const float* __restrict__ pw3, const float* __restrict__ pb3,
    const float* __restrict__ ww1, const float* __restrict__ wb1,
    const float* __restrict__ ww2, const float* __restrict__ wb2,
    const float* __restrict__ ww3, const float* __restrict__ wb3,
    _Float16* __restrict__ wtsT){
  __shared__ _Float16 st[8][48*NS];
  int tid = threadIdx.x, w = tid>>5, l = tid&31;
  int bp = blockIdx.x*8 + w;
  const float* g = gxyz + ((size_t)bp*NS + l)*3;
  float h0[3] = {g[0], g[1], g[2]};
  float h1[8];
#pragma unroll
  for (int o=0;o<8;o++){
    float s = pb1[o] + pw1[o*3+0]*h0[0] + pw1[o*3+1]*h0[1] + pw1[o*3+2]*h0[2];
    h1[o] = lrelu(s);
  }
  float h2[16];
#pragma unroll
  for (int o=0;o<16;o++){
    float s = pb2[o];
#pragma unroll
    for (int i=0;i<8;i++) s += pw2[o*8+i]*h1[i];
    h2[o] = lrelu(s);
  }
  float h3[16];
#pragma unroll
  for (int o=0;o<16;o++){
    float s = pb3[o];
#pragma unroll
    for (int i=0;i<16;i++) s += pw3[o*16+i]*h2[i];
    h3[o] = lrelu(s);
  }
  float h4[16];
#pragma unroll
  for (int o=0;o<16;o++){
    float s = wb1[o];
#pragma unroll
    for (int i=0;i<16;i++) s += ww1[o*16+i]*h3[i];
    h4[o] = lrelu(s);
  }
  float h5[32];
#pragma unroll
  for (int o=0;o<32;o++){
    float s = wb2[o];
#pragma unroll
    for (int i=0;i<16;i++) s += ww2[o*16+i]*h4[i];
    h5[o] = lrelu(s);
  }
  float wv[36];
#pragma unroll
  for (int o=0;o<36;o++){
    float s = wb3[o];
#pragma unroll
    for (int i=0;i<32;i++) s += ww3[o*32+i]*h5[i];
    wv[o] = s;                               // no activation
  }
  float s1 = 0.f;
#pragma unroll
  for (int m=0;m<36;m++) s1 += wv[m]*wv[m];
  s1 = sqrtf(fmaxf(s1, 1e-8f));
#pragma unroll
  for (int m=0;m<36;m++){
    float v = wv[m]*wv[m];
    v += __shfl_xor(v,16,32); v += __shfl_xor(v,8,32);
    v += __shfl_xor(v,4,32);  v += __shfl_xor(v,2,32); v += __shfl_xor(v,1,32);
    float s2 = fmaxf(sqrtf(fmaxf(v, 1e-8f)), 1.0f);
    wv[m] = wv[m] / (s1*s2);
  }
#pragma unroll
  for (int m=0;m<48;m++)
    st[w][m*NS + l] = (m < 36) ? (_Float16)wv[m] : (_Float16)0.0f;
  __syncthreads();
  unsigned* src = (unsigned*)&st[w][0];
  unsigned* dst = (unsigned*)(wtsT + (size_t)bp*48*NS);
  for (int i=l;i<48*NS/2;i+=32) dst[i] = src[i];
}

// ---------------------------------------------------------------------------
// K4: fused pf + conv1 + conv2 + conv3. One wave per workgroup, 4 samples/wave.
//   Per-wave LDS (f16):
//     img  [4][37*128] = 18944   (6x6 HWC map; row 36 = dump row)
//     bufA [4][16*128] =  8192   (4x4 conv1 out; stage-1 alias: gfT 128x40)
//     out2 [4][ 4*128] =  2048   (2x2 conv2 out)
//     wbuf [48*32]     =  1536   (TDM-loaded wts tile)
//   total 30720 f16 = 61,440 B
// ---------------------------------------------------------------------------
__global__ __launch_bounds__(32,1) void k_fused(
    const _Float16* __restrict__ featT, const _Float16* __restrict__ wtsT,
    const int* __restrict__ idxg,
    const _Float16* __restrict__ w1B, const _Float16* __restrict__ w2B,
    const _Float16* __restrict__ w3B,
    const float* __restrict__ pfb, const float* __restrict__ c1b,
    const float* __restrict__ c2b, const float* __restrict__ c3b,
    float* __restrict__ outg)
{
  extern __shared__ _Float16 sm[];
  int l  = threadIdx.x & 31;
  int lh = l >> 4, ll = l & 15;
  _Float16* img  = sm;                       // [4][IMGP]
  _Float16* bufA = sm + 4*IMGP;              // [4][2048]
  _Float16* out2 = sm + 4*IMGP + 4*2048;     // [4][512]
  _Float16* wbuf = sm + 4*IMGP + 4*2048 + 4*512; // [1536]
  _Float16* gfT  = bufA;                     // 128 x GFP (stage-1 only)
  int base = blockIdx.x * 4;

  // ------- Stage 1: pf^T(c,m) = gfT(c,s) @ wts(s,m), then img[m][c] ---------
  for (int t=0;t<4;t++){
    int bp = base + t;
    int b  = bp >> 11;
    __syncthreads();                         // gfT/wbuf safe to overwrite
    int myidx = idxg[(size_t)bp*NS + l];     // lane = neighbor sample s
    // DMA the 48x32 f16 wts tile into LDS, overlapped with the gather.
    tdm_load_1d((unsigned)(size_t)wbuf, wtsT + (size_t)bp*48*NS, 48*NS);
    // gather + transpose: lane owns row s=l of gf; scatter into gfT[c][s]
    // (stores are 2B lane-contiguous -> conflict-free)
    const _Float16* frow = featT + ((size_t)b*NPTS + myidx)*CINCH;
#pragma unroll
    for (int i=0;i<16;i++){
      v8h row = *(const v8h*)(frow + i*8);
#pragma unroll
      for (int q=0;q<8;q++) gfT[(i*8+q)*GFP + l] = row[q];
    }
    tdm_wait();                              // TENSORcnt == 0
    __syncthreads();                         // gfT + wbuf visible
    // B fragments: wts rows are s-contiguous -> two 16B runs at +lh*8 / +16+lh*8
    v16h bfr[3];
#pragma unroll
    for (int nt=0;nt<3;nt++){
      const _Float16* p = wbuf + (nt*16 + ll)*NS + lh*8;
      bfr[nt] = pack16(p, p+16);
    }
    _Float16* imgS = img + t*IMGP;
#pragma unroll
    for (int mt=0;mt<8;mt++){
      const _Float16* ap = gfT + (mt*16 + ll)*GFP + lh*8;
      v16h af = pack16(ap, ap+16);
      int cbase = mt*16 + lh*8;
      float bb[8];
#pragma unroll
      for (int r=0;r<8;r++) bb[r] = pfb[cbase+r];
#pragma unroll
      for (int nt=0;nt<3;nt++){
        v8f acc = {};
        acc = wmma16(af, bfr[nt], acc);
        v8h hv;
#pragma unroll
        for (int r=0;r<8;r++) hv[r] = (_Float16)lrelu(acc[r] + bb[r]);
        int m  = nt*16 + ll;
        int ms = m < 36 ? m : 36;            // clamp to dump row (no exec branch)
        *(v8h*)(imgS + ms*CINCH + cbase) = hv;
      }
    }
  }
  __syncthreads();

  // ------- Stage 2: conv1 3x3, 6x6 -> 4x4 (M=16 pos, 4 samples batched) -----
  {
    v8f acc[4][8] = {};
    int oy = ll >> 2, ox = ll & 3;
    for (int kk=0;kk<36;kk++){
      int tap  = kk >> 2;
      int py   = tap/3, px = tap - py*3;
      int cinb = (kk & 3)*32;
      int pix  = (oy+py)*6 + (ox+px);
      v16h af[4];
#pragma unroll
      for (int t=0;t<4;t++){
        const _Float16* ap = img + t*IMGP + pix*CINCH + cinb + lh*8;
        af[t] = pack16(ap, ap+16);
      }
      if (kk < 35)
        __builtin_prefetch(w1B + (((size_t)(kk+1)*8)*32)*16, 0, 0);
#pragma unroll
      for (int nt=0;nt<8;nt++){
        v16h bf = *(const v16h*)(w1B + (((size_t)kk*8 + nt)*32 + l)*16);
#pragma unroll
        for (int t=0;t<4;t++) acc[t][nt] = wmma16(af[t], bf, acc[t][nt]);
      }
    }
    __syncthreads();                    // img reads done; bufA may be overwritten
#pragma unroll
    for (int nt=0;nt<8;nt++){
      float bc = c1b[nt*16 + ll];
#pragma unroll
      for (int t=0;t<4;t++){
#pragma unroll
        for (int r=0;r<8;r++){
          int m = lh*8 + r;             // output position oy*4+ox
          bufA[t*2048 + m*CINCH + nt*16 + ll] = (_Float16)lrelu(acc[t][nt][r] + bc);
        }
      }
    }
  }
  __syncthreads();

  // ------- Stage 3: conv2 3x3, 4x4 -> 2x2 (M rows = sample*4 + pos) ---------
  {
    v8f acc[8] = {};
    int ts = ll >> 2, pos = ll & 3;
    int oy = pos >> 1, ox = pos & 1;
    for (int kk=0;kk<36;kk++){
      int tap  = kk >> 2;
      int py   = tap/3, px = tap - py*3;
      int cinb = (kk & 3)*32;
      int pix  = (oy+py)*4 + (ox+px);
      const _Float16* ap = bufA + ts*2048 + pix*CINCH + cinb + lh*8;
      v16h af = pack16(ap, ap+16);
#pragma unroll
      for (int nt=0;nt<8;nt++){
        v16h bf = *(const v16h*)(w2B + (((size_t)kk*8 + nt)*32 + l)*16);
        acc[nt] = wmma16(af, bf, acc[nt]);
      }
    }
    __syncthreads();
#pragma unroll
    for (int nt=0;nt<8;nt++){
      float bc = c2b[nt*16 + ll];
#pragma unroll
      for (int r=0;r<8;r++){
        int m = lh*8 + r;
        int tt = m >> 2, pp = m & 3;
        out2[tt*512 + pp*CINCH + nt*16 + ll] = (_Float16)lrelu(acc[nt][r] + bc);
      }
    }
  }
  __syncthreads();

  // ------- Stage 4: conv3 2x2 -> 1x1, C=128 -> 256 (M rows 0..3 used) -------
  {
    v8f acc[16] = {};
    for (int kk=0;kk<16;kk++){
      int pos  = kk >> 2;               // input pixel = kh*2+kw
      int cinb = (kk & 3)*32;
      v16h af = {};
      if (ll < 4){
        const _Float16* ap = out2 + ll*512 + pos*CINCH + cinb + lh*8;
        af = pack16(ap, ap+16);
      }
#pragma unroll
      for (int nt=0;nt<16;nt++){
        v16h bf = *(const v16h*)(w3B + (((size_t)kk*16 + nt)*32 + l)*16);
        acc[nt] = wmma16(af, bf, acc[nt]);
      }
    }
#pragma unroll
    for (int nt=0;nt<16;nt++){
      int co = nt*16 + ll;
      float bc = c3b[co];
#pragma unroll
      for (int r=0;r<8;r++){
        int m = lh*8 + r;
        if (m < 4){
          int bp = base + m;
          int b = bp >> 11, p = bp & (NQ-1);
          outg[(size_t)BPTOT*3 + ((size_t)b*COCH + co)*NQ + p] = lrelu(acc[nt][r] + bc);
        }
      }
    }
  }
}

// ---------------------------------------------------------------------------
extern "C" void kernel_launch(void* const* d_in, const int* in_sizes, int n_in,
                              void* d_out, int out_size, void* d_ws, size_t ws_size,
                              hipStream_t stream)
{
  const float* xyz      = (const float*)d_in[0];
  const float* features = (const float*)d_in[1];
  const float* new_xyz  = (const float*)d_in[2];
  const float* bias     = (const float*)d_in[3];
  const float* pw1 = (const float*)d_in[4];  const float* pb1 = (const float*)d_in[5];
  const float* pw2 = (const float*)d_in[6];  const float* pb2 = (const float*)d_in[7];
  const float* pw3 = (const float*)d_in[8];  const float* pb3 = (const float*)d_in[9];
  const float* ww1 = (const float*)d_in[10]; const float* wb1 = (const float*)d_in[11];
  const float* ww2 = (const float*)d_in[12]; const float* wb2 = (const float*)d_in[13];
  const float* ww3 = (const float*)d_in[14]; const float* wb3 = (const float*)d_in[15];
  const float* c1w = (const float*)d_in[16]; const float* c1b = (const float*)d_in[17];
  const float* c2w = (const float*)d_in[18]; const float* c2b = (const float*)d_in[19];
  const float* c3w = (const float*)d_in[20]; const float* c3b = (const float*)d_in[21];

  char* ws = (char*)d_ws;
  _Float16* featTb = (_Float16*)(ws + 0);          //  8,388,608 B
  _Float16* wtsT   = (_Float16*)(ws + 8388608);    // 25,165,824 B
  int*      idxg   = (int*)     (ws + 33554432);   //  1,048,576 B
  float*    gxyz   = (float*)   (ws + 34603008);   //  3,145,728 B
  _Float16* w1B    = (_Float16*)(ws + 37748736);   //    294,912 B
  _Float16* w2B    = (_Float16*)(ws + 38043648);   //    294,912 B
  _Float16* w3B    = (_Float16*)(ws + 38338560);   //    262,144 B  (total ~38.6 MB)
  float*    outf   = (float*)d_out;

  // output head: new_xyz passthrough
  hipMemcpyAsync(d_out, d_in[2], (size_t)BPTOT*3*sizeof(float),
                 hipMemcpyDeviceToDevice, stream);

  k_featT    <<<dim3(NPTS/32, CINCH/32, NB), dim3(32,8), 0, stream>>>(features, featTb);
  k_wprep    <<<576, 256, 0, stream>>>(c1w, c2w, c3w, w1B, w2B, w3B);
  k_ballquery<<<BPTOT/8, 256, 0, stream>>>(xyz, new_xyz, idxg, gxyz);
  k_mlp      <<<BPTOT/8, 256, 0, stream>>>(gxyz, pw1,pb1, pw2,pb2, pw3,pb3,
                                           ww1,wb1, ww2,wb2, ww3,wb3, wtsT);
  k_fused    <<<BPTOT/4, 32, (4*IMGP + 4*2048 + 4*512 + 48*NS)*sizeof(_Float16),
               stream>>>(featTb, wtsT, idxg, w1B, w2B, w3B, bias,
                         c1b, c2b, c3b, outf);
}
